// AttnDecoderRNN_7275674600338
// MI455X (gfx1250) — matrix operand order repaired and checked by reading, not compiled, not run on hardware
//
#include <hip/hip_runtime.h>
#include <hip/hip_bf16.h>

#define Hh 1024
#define Bb 64
#define Vv 32000
#define NSTEP 50

typedef __attribute__((ext_vector_type(16))) __bf16 v16bf;
typedef __attribute__((ext_vector_type(8)))  float  v8f;

union TileU { uint4 q[2]; v16bf v; };

__device__ __forceinline__ unsigned short f2bf(float f) {
  unsigned u = __float_as_uint(f);
  unsigned r = u + 0x7FFFu + ((u >> 16) & 1u);   // round-to-nearest-even
  return (unsigned short)(r >> 16);
}

// ---- WMMA operand loaders -------------------------------------------------
// A (16x32 bf16, row-major source, ld=Hh):
// lanes 0-15: M=lane, K={0..15}; lanes 16-31: M=lane-16, K={16..31}
__device__ __forceinline__ v16bf load_a16(const unsigned short* __restrict__ A,
                                          int row0, int kbase, int lane) {
  int row = row0 + (lane & 15);
  int kk  = kbase + (lane & 16);  // +16 for upper half-wave
  TileU u;
  u.q[0] = *reinterpret_cast<const uint4*>(A + (size_t)row * Hh + kk);
  u.q[1] = *reinterpret_cast<const uint4*>(A + (size_t)row * Hh + kk + 8);
  return u.v;
}

// B (32x16 bf16) from pre-packed tile: lane-contiguous 16 elements (32 bytes)
__device__ __forceinline__ v16bf load_b16(const unsigned short* __restrict__ Bt, int lane) {
  TileU u;
  const unsigned short* p = Bt + lane * 16;
  u.q[0] = *reinterpret_cast<const uint4*>(p);
  u.q[1] = *reinterpret_cast<const uint4*>(p + 8);
  return u.v;
}

__device__ __forceinline__ v8f wmma_bf16(v16bf a, v16bf b, v8f c) {
  // (neg_a, A, neg_b, B, c_mod, C, reuse_a, reuse_b)
  return __builtin_amdgcn_wmma_f32_16x16x32_bf16(false, a, false, b, (short)0, c, false, false);
}

__device__ __forceinline__ unsigned long long make_key(float val, int v) {
  unsigned u = __float_as_uint(val);
  u = (u & 0x80000000u) ? ~u : (u | 0x80000000u);  // monotonic float ordering
  // low word: larger for smaller v  ->  ties break to first (lowest) index
  return ((unsigned long long)u << 32) | (unsigned long long)(0xFFFFFFFFu - (unsigned)v);
}

// ---- Weight packing: fp32 [K][N] row-major -> bf16 WMMA B-operand tiles ---
// tile (nt,kt): 512 bf16; lane = n%16 + 16*(k_in_tile/16); elem j = k_in_tile%16
__global__ void pack_b_kernel(const float* __restrict__ W, unsigned short* __restrict__ P,
                              int K, int N) {
  long long idx = (long long)blockIdx.x * blockDim.x + threadIdx.x;
  if (idx >= (long long)K * N) return;
  int n = (int)(idx % N);
  int k = (int)(idx / N);
  int kt = k >> 5, kk = k & 31;
  int nt = n >> 4, nn = n & 15;
  int lane = nn + ((kk >> 4) << 4);
  int j    = kk & 15;
  size_t dst = (((size_t)nt * (K >> 5) + kt) << 9) + (size_t)lane * 16 + j;
  P[dst] = f2bf(W[(size_t)k * N + n]);
}

// ---- Init: tok=SOS, active=1, h/c copy, x = emb[SOS] ----------------------
__global__ void init_kernel(const float* __restrict__ h_in, const float* __restrict__ c_in,
                            const float* __restrict__ emb,
                            float* __restrict__ h, float* __restrict__ c,
                            unsigned short* __restrict__ hbf, unsigned short* __restrict__ xbf,
                            int* __restrict__ tok, int* __restrict__ active) {
  int i = blockIdx.x * blockDim.x + threadIdx.x;
  if (i >= Bb * Hh) return;
  float hv = h_in[i];
  h[i]   = hv;
  hbf[i] = f2bf(hv);
  c[i]   = c_in[i];
  xbf[i] = f2bf(emb[(size_t)1 * Hh + (i & (Hh - 1))]);  // SOS_TOKEN == 1
  if (i == 0) *active = 1;
  if (i < Bb) tok[i] = 1;
}

// ---- Fused gates GEMM + LSTM pointwise ------------------------------------
// 32 blocks x 128 threads. Wave w = gate w; block nb covers columns
// [nb*32, nb*32+32). Each wave: acc[4 mt][2 nt]. Gate values staged in LDS,
// then the block applies the LSTM nonlinearity for its 64x32 slab.
__global__ void gates_lstm_kernel(const unsigned short* __restrict__ xbf,
                                  const unsigned short* __restrict__ hbf,
                                  const unsigned short* __restrict__ pwih,
                                  const unsigned short* __restrict__ pwhh,
                                  const float* __restrict__ b_ih,
                                  const float* __restrict__ b_hh,
                                  const float* __restrict__ c,
                                  float* __restrict__ hnew, float* __restrict__ cnew,
                                  unsigned short* __restrict__ hnbf,
                                  unsigned long long* __restrict__ amax) {
  __shared__ float s_g[4][Bb][32];
  int tid  = threadIdx.x;
  int lane = tid & 31;
  int g    = tid >> 5;          // wave index == gate index
  int nb   = blockIdx.x;        // 0..31

  if (nb == 0 && tid < Bb) amax[tid] = 0ull;  // reset argmax keys for this step

  v8f zero = {0.f,0.f,0.f,0.f,0.f,0.f,0.f,0.f};
  v8f acc[4][2] = {{zero,zero},{zero,zero},{zero,zero},{zero,zero}};

  const size_t ntBlk = (size_t)(Hh / 32) << 9;  // elems per nt column-block (32KB)
  const unsigned short* pB = pwih + (size_t)g * Hh * Hh + (size_t)(nb * 2) * ntBlk;
  for (int kt = 0; kt < Hh / 32; ++kt) {
    const unsigned short* b0 = pB + ((size_t)kt << 9);
    v16bf bt0 = load_b16(b0, lane);
    v16bf bt1 = load_b16(b0 + ntBlk, lane);
    if (kt + 1 < Hh / 32) __builtin_prefetch(b0 + 512, 0, 1);
    #pragma unroll
    for (int mt = 0; mt < 4; ++mt) {
      v16bf at = load_a16(xbf, mt * 16, kt * 32, lane);
      acc[mt][0] = wmma_bf16(at, bt0, acc[mt][0]);
      acc[mt][1] = wmma_bf16(at, bt1, acc[mt][1]);
    }
  }
  pB = pwhh + (size_t)g * Hh * Hh + (size_t)(nb * 2) * ntBlk;
  for (int kt = 0; kt < Hh / 32; ++kt) {
    const unsigned short* b0 = pB + ((size_t)kt << 9);
    v16bf bt0 = load_b16(b0, lane);
    v16bf bt1 = load_b16(b0 + ntBlk, lane);
    if (kt + 1 < Hh / 32) __builtin_prefetch(b0 + 512, 0, 1);
    #pragma unroll
    for (int mt = 0; mt < 4; ++mt) {
      v16bf at = load_a16(hbf, mt * 16, kt * 32, lane);
      acc[mt][0] = wmma_bf16(at, bt0, acc[mt][0]);
      acc[mt][1] = wmma_bf16(at, bt1, acc[mt][1]);
    }
  }

  int rbase = (lane >> 4) << 3;
  #pragma unroll
  for (int j = 0; j < 2; ++j) {
    int n    = (nb * 2 + j) * 16 + (lane & 15);
    int cc   = j * 16 + (lane & 15);
    float bias = b_ih[g * Hh + n] + b_hh[g * Hh + n];
    #pragma unroll
    for (int mt = 0; mt < 4; ++mt) {
      #pragma unroll
      for (int r = 0; r < 8; ++r) {
        s_g[g][mt * 16 + rbase + r][cc] = acc[mt][j][r] + bias;
      }
    }
  }
  __syncthreads();

  for (int e = tid; e < Bb * 32; e += blockDim.x) {
    int row = e >> 5, cc = e & 31;
    int i = row * Hh + nb * 32 + cc;
    float ig = 1.f / (1.f + __expf(-s_g[0][row][cc]));
    float fg = 1.f / (1.f + __expf(-s_g[1][row][cc]));
    float cg = tanhf(s_g[2][row][cc]);
    float og = 1.f / (1.f + __expf(-s_g[3][row][cc]));
    float cn = fg * c[i] + ig * cg;
    float hn = og * tanhf(cn);
    cnew[i] = cn;
    hnew[i] = hn;
    hnbf[i] = f2bf(hn);
  }
}

// ---- Logits GEMM + fused per-row argmax -----------------------------------
// 250 blocks x 128 threads; wave -> pair of 16-wide vocab tiles (32 columns).
__global__ void logits_kernel(const unsigned short* __restrict__ hnbf,
                              const unsigned short* __restrict__ poutw,
                              const float* __restrict__ out_b,
                              const float* __restrict__ std_all, int t,
                              unsigned long long* __restrict__ amax) {
  int lane = threadIdx.x & 31;
  int ntp  = blockIdx.x * (blockDim.x >> 5) + (threadIdx.x >> 5);  // 0..999
  v8f zero = {0.f,0.f,0.f,0.f,0.f,0.f,0.f,0.f};
  v8f acc[4][2] = {{zero,zero},{zero,zero},{zero,zero},{zero,zero}};

  const size_t ntBlk = (size_t)(Hh / 32) << 9;
  const unsigned short* pB = poutw + (size_t)(ntp * 2) * ntBlk;
  for (int kt = 0; kt < Hh / 32; ++kt) {
    const unsigned short* b0 = pB + ((size_t)kt << 9);
    v16bf bt0 = load_b16(b0, lane);
    v16bf bt1 = load_b16(b0 + ntBlk, lane);
    if (kt + 1 < Hh / 32) __builtin_prefetch(b0 + 512, 0, 1);
    #pragma unroll
    for (int mt = 0; mt < 4; ++mt) {
      v16bf at = load_a16(hnbf, mt * 16, kt * 32, lane);
      acc[mt][0] = wmma_bf16(at, bt0, acc[mt][0]);
      acc[mt][1] = wmma_bf16(at, bt1, acc[mt][1]);
    }
  }

  int v0 = (ntp * 2) * 16 + (lane & 15);
  int v1 = v0 + 16;
  float ob0 = out_b[v0];
  float ob1 = out_b[v1];
  const float* sp = std_all + (size_t)t * Bb * Vv;
  int rbase = (lane >> 4) << 3;
  #pragma unroll
  for (int mt = 0; mt < 4; ++mt) {
    #pragma unroll
    for (int r = 0; r < 8; ++r) {
      int row = mt * 16 + rbase + r;
      const float* srow = sp + (size_t)row * Vv;
      float a0 = acc[mt][0][r] + ob0 + srow[v0];
      float a1 = acc[mt][1][r] + ob1 + srow[v1];
      unsigned long long k0 = make_key(a0, v0);
      unsigned long long k1 = make_key(a1, v1);
      unsigned long long key = k0 > k1 ? k0 : k1;
      #pragma unroll
      for (int m = 1; m <= 8; m <<= 1) {  // reduce over 16 columns (half-wave)
        unsigned lo = __shfl_xor((unsigned)key, m, 32);
        unsigned hi = __shfl_xor((unsigned)(key >> 32), m, 32);
        unsigned long long o = ((unsigned long long)hi << 32) | (unsigned long long)lo;
        key = key > o ? key : o;
      }
      if ((lane & 15) == 0) atomicMax(&amax[row], key);
    }
  }
}

// ---- Finalize step: tokens, active flag, h/c commit, next-x gather --------
__global__ void finalize_kernel(const unsigned long long* __restrict__ amax,
                                int* __restrict__ tok, int* __restrict__ active,
                                const float* __restrict__ hnew, const float* __restrict__ cnew,
                                float* __restrict__ h, float* __restrict__ c,
                                unsigned short* __restrict__ hbf, unsigned short* __restrict__ xbf,
                                const float* __restrict__ emb,
                                float* __restrict__ out_tok, float* __restrict__ out_h) {
  __shared__ int s_act, s_maxtok;
  int tid = threadIdx.x;
  if (tid == 0) { s_act = *active; s_maxtok = 0; }
  __syncthreads();
  int act = s_act;
  if (tid < Bb) {
    unsigned long long key = amax[tid];
    int tn = (int)(0xFFFFFFFFu - (unsigned)(key & 0xFFFFFFFFull));
    tok[tid] = act ? tn : tok[tid];
    out_tok[tid] = act ? (float)tn : 0.0f;
    atomicMax(&s_maxtok, tn);
  }
  __syncthreads();
  if (tid == 0) *active = (act && s_maxtok > 0) ? 1 : 0;  // EOS_TOKEN == 0
  for (int i = tid; i < Bb * Hh; i += blockDim.x) {
    float hv = act ? hnew[i] : h[i];
    float cv = act ? cnew[i] : c[i];
    h[i] = hv; c[i] = cv;
    hbf[i] = f2bf(hv);
    out_h[i] = hv;
    int b = i >> 10, k = i & (Hh - 1);
    xbf[i] = f2bf(emb[(size_t)tok[b] * Hh + k]);
  }
}

extern "C" void kernel_launch(void* const* d_in, const int* in_sizes, int n_in,
                              void* d_out, int out_size, void* d_ws, size_t ws_size,
                              hipStream_t stream) {
  (void)in_sizes; (void)n_in; (void)out_size; (void)ws_size;
  const float* std_in = (const float*)d_in[1];
  const float* h_in   = (const float*)d_in[2];
  const float* c_in   = (const float*)d_in[3];
  const float* emb    = (const float*)d_in[4];
  const float* w_ih   = (const float*)d_in[5];
  const float* w_hh   = (const float*)d_in[6];
  const float* b_ih   = (const float*)d_in[7];
  const float* b_hh   = (const float*)d_in[8];
  const float* out_w  = (const float*)d_in[9];
  const float* out_b  = (const float*)d_in[10];

  // workspace carve-up (256B aligned)
  char* base = (char*)d_ws;
  size_t off = 0;
  auto carve = [&](size_t bytes) -> void* {
    void* p = base + off;
    off = (off + bytes + 255) & ~(size_t)255;
    return p;
  };
  unsigned short* pwih = (unsigned short*)carve((size_t)4 * Hh * Hh * 2);  // 8 MB
  unsigned short* pwhh = (unsigned short*)carve((size_t)4 * Hh * Hh * 2);  // 8 MB
  unsigned short* pout = (unsigned short*)carve((size_t)Hh * Vv * 2);      // 62.5 MB
  unsigned short* xbf  = (unsigned short*)carve((size_t)Bb * Hh * 2);
  unsigned short* hbf  = (unsigned short*)carve((size_t)Bb * Hh * 2);
  unsigned short* hnbf = (unsigned short*)carve((size_t)Bb * Hh * 2);
  float* hF  = (float*)carve((size_t)Bb * Hh * 4);
  float* cF  = (float*)carve((size_t)Bb * Hh * 4);
  float* hnF = (float*)carve((size_t)Bb * Hh * 4);
  float* cnF = (float*)carve((size_t)Bb * Hh * 4);
  unsigned long long* amax = (unsigned long long*)carve((size_t)Bb * 8);
  int* tok    = (int*)carve((size_t)Bb * 4);
  int* active = (int*)carve(16);

  // ---- one-time packing (re-done every call; deterministic) ----
  {
    int blk = 256;
    int nb  = (Hh * Hh + blk - 1) / blk;
    for (int g = 0; g < 4; ++g) {
      pack_b_kernel<<<nb, blk, 0, stream>>>(w_ih + (size_t)g * Hh * Hh,
                                            pwih + (size_t)g * Hh * Hh, Hh, Hh);
      pack_b_kernel<<<nb, blk, 0, stream>>>(w_hh + (size_t)g * Hh * Hh,
                                            pwhh + (size_t)g * Hh * Hh, Hh, Hh);
    }
    long long e = (long long)Hh * Vv;
    int nb2 = (int)((e + blk - 1) / blk);
    pack_b_kernel<<<nb2, blk, 0, stream>>>(out_w, pout, Hh, Vv);
  }
  init_kernel<<<(Bb * Hh + 255) / 256, 256, 0, stream>>>(h_in, c_in, emb, hF, cF,
                                                         hbf, xbf, tok, active);

  float* out_tokens = (float*)d_out;              // [50][64]
  float* out_hfinal = (float*)d_out + NSTEP * Bb; // [64][1024]

  for (int t = 0; t < NSTEP; ++t) {
    gates_lstm_kernel<<<32, 128, 0, stream>>>(xbf, hbf, pwih, pwhh, b_ih, b_hh,
                                              cF, hnF, cnF, hnbf, amax);
    logits_kernel<<<250, 128, 0, stream>>>(hnbf, pout, out_b, std_in, t, amax);
    finalize_kernel<<<1, 256, 0, stream>>>(amax, tok, active, hnF, cnF, hF, cF,
                                           hbf, xbf, emb, out_tokens + (size_t)t * Bb,
                                           out_hfinal);
  }
}
